// hypercorre_topk2_69183333204399
// MI455X (gfx1250) — matrix operand order repaired and checked by reading, not compiled, not run on hardware
//
#include <hip/hip_runtime.h>
#include <math.h>

// ---------------------------------------------------------------------------
// CDNA5 (gfx1250) implementation of the HyperCorre/top-k pipeline.
// Matmuls use v_wmma_f32_16x16x32_bf16 (bf16 mul, f32 accumulate), wave32.
// Each wave computes a 16x64 output strip (4 WMMA tiles) so the A fragment
// is loaded once per K-slab and reused 4x; row-major fragment operands are
// loaded as float4 (global_load_b128). Pipeline is HBM-bound (~700MB traffic
// vs ~16 GFLOP at 23.3 TB/s), so minimizing bytes/FLOP is the target.
// ---------------------------------------------------------------------------

typedef __attribute__((ext_vector_type(16))) __bf16 v16bf;
typedef __attribute__((ext_vector_type(8)))  float  v8f;

__device__ __forceinline__ __bf16 f2bf(float f) { return (__bf16)f; }

// Pack 4 contiguous float4s (16 floats, ascending K) into a v16bf fragment
// whose element order matches "pair (2v,2v+1) = consecutive K" packing.
__device__ __forceinline__ v16bf cvt16(float4 a, float4 b, float4 c, float4 d) {
  v16bf r;
  r[0]  = f2bf(a.x); r[1]  = f2bf(a.y); r[2]  = f2bf(a.z); r[3]  = f2bf(a.w);
  r[4]  = f2bf(b.x); r[5]  = f2bf(b.y); r[6]  = f2bf(b.z); r[7]  = f2bf(b.w);
  r[8]  = f2bf(c.x); r[9]  = f2bf(c.y); r[10] = f2bf(c.z); r[11] = f2bf(c.w);
  r[12] = f2bf(d.x); r[13] = f2bf(d.y); r[14] = f2bf(d.z); r[15] = f2bf(d.w);
  return r;
}

// A fragment (16x32, 16-bit), row-major source: lane covers K runs
// [half*8, half*8+8) and [16+half*8, 16+half*8+8).  p = row base + k0.
__device__ __forceinline__ v16bf frag_a_rowmajor(const float* p, int half) {
  const float4* q = reinterpret_cast<const float4*>(p);
  return cvt16(q[half * 2], q[half * 2 + 1], q[4 + half * 2], q[5 + half * 2]);
}

// B fragment (32x16, 16-bit): lane covers K runs
// [half*8, half*8+8) and [16+half*8, 16+half*8+8).
__device__ __forceinline__ v16bf frag_b(const float* p, int half) {
  const float4* q = reinterpret_cast<const float4*>(p);
  return cvt16(q[half * 2], q[half * 2 + 1], q[4 + half * 2], q[5 + half * 2]);
}

// A-fragment (16x32) per-lane K index for strided (column-major) sources:
// lanes 0-15 K={0..7,16..23}, lanes 16-31 K={8..15,24..31}; 2 values/VGPR.
__device__ __forceinline__ int a_kbase(int v, int half) {
  return ((v < 4) ? 0 : 16) + half * 8 + ((v & 3) << 1);
}

// ---------------------------------------------------------------------------
// y[g, t, o] = sum_c x[g, c, t] * w[o, c] + bias[o]
// A read transposed from NCHW (strided); one wave -> 16 tokens x 64 channels.
// grid = (T/16, O/64, G), block = 32
// ---------------------------------------------------------------------------
__global__ void proj_gemm_wmma(const float* __restrict__ x,
                               const float* __restrict__ w,
                               const float* __restrict__ bias,
                               float* __restrict__ y,
                               int C, int T, int O) {
  const int tt   = blockIdx.x;
  const int to0  = blockIdx.y * 4;
  const int g    = blockIdx.z;
  const int lane = threadIdx.x;
  const int half = lane >> 4;
  const int lo   = lane & 15;

  const float* xg = x + (size_t)g * C * T;
  const int t = tt * 16 + lo;

  const float* wrow[4];
#pragma unroll
  for (int ti = 0; ti < 4; ++ti)
    wrow[ti] = w + (size_t)((to0 + ti) * 16 + lo) * C;

  v8f acc[4] = {};
  for (int k0 = 0; k0 < C; k0 += 32) {
    if (k0 + 32 < C)
      __builtin_prefetch(&xg[(size_t)(k0 + 32) * T + t], 0, 0);
    v16bf a;
#pragma unroll
    for (int v = 0; v < 8; ++v) {
      const int ka = a_kbase(v, half);
      a[2 * v]     = f2bf(xg[(size_t)(k0 + ka) * T + t]);
      a[2 * v + 1] = f2bf(xg[(size_t)(k0 + ka + 1) * T + t]);
    }
#pragma unroll
    for (int ti = 0; ti < 4; ++ti) {
      const v16bf b = frag_b(wrow[ti] + k0, half);
      acc[ti] = __builtin_amdgcn_wmma_f32_16x16x32_bf16(false, a, false, b,
                                                        (short)0, acc[ti],
                                                        false, false);
    }
  }
#pragma unroll
  for (int ti = 0; ti < 4; ++ti) {
    const int o = (to0 + ti) * 16 + lo;
    const float bs = bias[o];
#pragma unroll
    for (int r = 0; r < 8; ++r) {
      const int m = tt * 16 + half * 8 + r;
      y[((size_t)g * T + m) * O + o] = acc[ti][r] + bs;
    }
  }
}

// ---------------------------------------------------------------------------
// out[g, m, n] = sum_c sq[b, m, c] * qk[g, n, c],  b = g / nq
// One wave -> 16 rows x 64 cols.  grid = (N/64, M/16, G), block = 32
// ---------------------------------------------------------------------------
__global__ void corr_full_wmma(const float* __restrict__ sq,
                               const float* __restrict__ qk,
                               float* __restrict__ out,
                               int M, int N, int C, int nq) {
  const int tn0 = blockIdx.x * 4, tm = blockIdx.y, g = blockIdx.z;
  const int b    = g / nq;
  const int lane = threadIdx.x;
  const int half = lane >> 4;
  const int lo   = lane & 15;

  const float* Ap = sq + ((size_t)b * M + (size_t)tm * 16 + lo) * C;
  const float* Bp[4];
#pragma unroll
  for (int ti = 0; ti < 4; ++ti)
    Bp[ti] = qk + ((size_t)g * N + (size_t)(tn0 + ti) * 16 + lo) * C;

  v8f acc[4] = {};
  for (int k0 = 0; k0 < C; k0 += 32) {
    if (k0 + 32 < C) __builtin_prefetch(&Ap[k0 + 32], 0, 0);
    const v16bf a = frag_a_rowmajor(Ap + k0, half);
#pragma unroll
    for (int ti = 0; ti < 4; ++ti) {
      const v16bf bb = frag_b(Bp[ti] + k0, half);
      acc[ti] = __builtin_amdgcn_wmma_f32_16x16x32_bf16(false, a, false, bb,
                                                        (short)0, acc[ti],
                                                        false, false);
    }
  }
#pragma unroll
  for (int ti = 0; ti < 4; ++ti) {
    const int n = (tn0 + ti) * 16 + lo;
#pragma unroll
    for (int r = 0; r < 8; ++r) {
      const int m = tm * 16 + half * 8 + r;
      out[((size_t)g * M + m) * N + n] = acc[ti][r];
    }
  }
}

// ---------------------------------------------------------------------------
// Gathered correlation written in maps() layout:
//   out[(g*SS + j) * M + m] = sum_c sq[b, m, c] * qk[g, idx[g*SS+j], c]
// One wave -> 16 rows x 64 selected cols. grid = (SS/64, M/16, G), block = 32
// ---------------------------------------------------------------------------
__global__ void corr_gather_wmma(const float* __restrict__ sq,
                                 const float* __restrict__ qk,
                                 const int* __restrict__ idx,
                                 float* __restrict__ out,
                                 int M, int SS, int C, int NT, int nq) {
  const int tj0 = blockIdx.x * 4, tm = blockIdx.y, g = blockIdx.z;
  const int b    = g / nq;
  const int lane = threadIdx.x;
  const int half = lane >> 4;
  const int lo   = lane & 15;

  const float* Ap = sq + ((size_t)b * M + (size_t)tm * 16 + lo) * C;
  const float* Bp[4];
#pragma unroll
  for (int ti = 0; ti < 4; ++ti) {
    const int j    = (tj0 + ti) * 16 + lo;
    const int qrow = idx[(size_t)g * SS + j];
    Bp[ti] = qk + ((size_t)g * NT + qrow) * C;
  }

  v8f acc[4] = {};
  for (int k0 = 0; k0 < C; k0 += 32) {
    if (k0 + 32 < C) __builtin_prefetch(&Ap[k0 + 32], 0, 0);
    const v16bf a = frag_a_rowmajor(Ap + k0, half);
#pragma unroll
    for (int ti = 0; ti < 4; ++ti) {
      const v16bf bb = frag_b(Bp[ti] + k0, half);
      acc[ti] = __builtin_amdgcn_wmma_f32_16x16x32_bf16(false, a, false, bb,
                                                        (short)0, acc[ti],
                                                        false, false);
    }
  }
#pragma unroll
  for (int ti = 0; ti < 4; ++ti) {
    const int j = (tj0 + ti) * 16 + lo;
#pragma unroll
    for (int r = 0; r < 8; ++r) {
      const int m = tm * 16 + half * 8 + r;
      out[((size_t)g * SS + j) * M + m] = acc[ti][r];
    }
  }
}

// ---------------------------------------------------------------------------
// score[g, q] = sum of top-5 over s of att[g, s, q]
// ---------------------------------------------------------------------------
__global__ void topk5_sum_kernel(const float* __restrict__ att,
                                 float* __restrict__ score, int S, int Q) {
  const int g = blockIdx.y;
  const int q = blockIdx.x * blockDim.x + threadIdx.x;
  if (q >= Q) return;
  const float* p = att + (size_t)g * S * Q + q;
  float t0 = -3.4e38f, t1 = t0, t2 = t0, t3 = t0, t4 = t0;
  for (int s = 0; s < S; ++s) {
    const float x = p[(size_t)s * Q];
    if (x > t4) {
      if (x > t0)      { t4 = t3; t3 = t2; t2 = t1; t1 = t0; t0 = x; }
      else if (x > t1) { t4 = t3; t3 = t2; t2 = t1; t1 = x; }
      else if (x > t2) { t4 = t3; t3 = t2; t2 = x; }
      else if (x > t3) { t4 = t3; t3 = x; }
      else             { t4 = x; }
    }
  }
  score[(size_t)g * Q + q] = t0 + t1 + t2 + t3 + t4;
}

// ---------------------------------------------------------------------------
// Per group g: indices of the top-SS scores (ties -> lower index, like
// lax.top_k), emitted in ascending index order (== jnp.sort of top_k idx).
// Bitonic sort in LDS; block = Q threads, grid = G.
// ---------------------------------------------------------------------------
__device__ __forceinline__ bool kv_before(float va, int ia, float vb, int ib) {
  return (va > vb) || (va == vb && ia < ib);   // value desc, idx asc
}

__global__ void select_topk_sorted_kernel(const float* __restrict__ score,
                                          int* __restrict__ idx_out,
                                          int Q, int SS) {
  __shared__ float sv[1024];
  __shared__ int   si[1024];
  const int t = threadIdx.x;
  const int g = blockIdx.x;
  sv[t] = score[(size_t)g * Q + t];
  si[t] = t;
  __syncthreads();

  for (int k = 2; k <= Q; k <<= 1) {
    for (int j = k >> 1; j > 0; j >>= 1) {
      const int ixj = t ^ j;
      if (ixj > t) {
        const bool dir = ((t & k) == 0);
        const float va = sv[t], vb = sv[ixj];
        const int   ia = si[t], ib = si[ixj];
        const bool inorder = kv_before(va, ia, vb, ib);
        if (inorder != dir) {
          sv[t] = vb; si[t] = ib;
          sv[ixj] = va; si[ixj] = ia;
        }
      }
      __syncthreads();
    }
  }
  for (int k = 2; k <= SS; k <<= 1) {
    for (int j = k >> 1; j > 0; j >>= 1) {
      const int ixj = t ^ j;
      if (t < SS && ixj > t && ixj < SS) {
        const bool dir = ((t & k) == 0);
        const int ia = si[t], ib = si[ixj];
        if ((ia <= ib) != dir) { si[t] = ib; si[ixj] = ia; }
      }
      __syncthreads();
    }
  }
  if (t < SS) idx_out[(size_t)g * SS + t] = si[t];
}

// ---------------------------------------------------------------------------
// Fused conv2d + GroupNorm(1 group) + ReLU per sample, one workgroup per
// sample. Optional second operand `in2` [P,CI,IH2,IW2] is bilinearly resized
// (align_corners=True) to HxW during LDS staging and added (fuses the
// F.interpolate + add of the HPN learner — saves a full HBM round trip).
// block = 256, grid = P. CI*H*W <= 8192, CO*H*W <= 8192.
// ---------------------------------------------------------------------------
__global__ void conv_gn_relu_kernel(const float* __restrict__ in,
                                    const float* __restrict__ in2,
                                    int IH2, int IW2,
                                    const float* __restrict__ cw,
                                    const float* __restrict__ cb,
                                    const float* __restrict__ gw,
                                    const float* __restrict__ gb,
                                    float* __restrict__ out,
                                    int CI, int CO, int H, int W,
                                    int KS, int pad) {
  __shared__ float s_in[2 * 64 * 64];
  __shared__ float rs[256];
  __shared__ float rs2[256];

  const int p   = blockIdx.x;
  const int tid = threadIdx.x;
  const int nin = CI * H * W;
  const float* ip = in + (size_t)p * nin;
  const float* ip2 = in2 ? in2 + (size_t)p * CI * IH2 * IW2 : nullptr;

  for (int i = tid; i < nin; i += 256) {
    float v = ip[i];
    if (ip2) {
      if (IH2 == H && IW2 == W) {
        v += ip2[i];
      } else {
        const int x = i % W;
        int rem = i / W;
        const int y = rem % H;
        const int c = rem / H;
        const float sy = (float)(IH2 - 1) / (float)(H - 1);
        const float sx = (float)(IW2 - 1) / (float)(W - 1);
        const float fy = y * sy, fx = x * sx;
        const int y0 = (int)fy, x0 = (int)fx;
        const int y1 = (y0 + 1 < IH2) ? y0 + 1 : IH2 - 1;
        const int x1 = (x0 + 1 < IW2) ? x0 + 1 : IW2 - 1;
        const float wy = fy - (float)y0, wx = fx - (float)x0;
        const float* q = ip2 + (size_t)c * IH2 * IW2;
        const float top = q[y0 * IW2 + x0] * (1.f - wx) + q[y0 * IW2 + x1] * wx;
        const float bot = q[y1 * IW2 + x0] * (1.f - wx) + q[y1 * IW2 + x1] * wx;
        v += top * (1.f - wy) + bot * wy;
      }
    }
    s_in[i] = v;
  }
  __syncthreads();

  const int nout = CO * H * W;
  float vals[32];
  float s = 0.f, s2 = 0.f;
  int e = 0;
  for (int i = tid; i < nout; i += 256, ++e) {
    const int co  = i / (H * W);
    const int rem = i % (H * W);
    const int y = rem / W, x = rem % W;
    float acc = cb[co];
    for (int ci = 0; ci < CI; ++ci) {
      for (int ky = 0; ky < KS; ++ky) {
        const int yy = y + ky - pad;
        if (yy < 0 || yy >= H) continue;
        for (int kx = 0; kx < KS; ++kx) {
          const int xx = x + kx - pad;
          if (xx < 0 || xx >= W) continue;
          acc += s_in[(ci * H + yy) * W + xx] *
                 cw[((co * CI + ci) * KS + ky) * KS + kx];
        }
      }
    }
    vals[e] = acc;
    s += acc;
    s2 += acc * acc;
  }

  rs[tid] = s;
  rs2[tid] = s2;
  __syncthreads();
  for (int off = 128; off > 0; off >>= 1) {
    if (tid < off) { rs[tid] += rs[tid + off]; rs2[tid] += rs2[tid + off]; }
    __syncthreads();
  }
  const float inv_n = 1.f / (float)nout;
  const float mu   = rs[0] * inv_n;
  const float var  = rs2[0] * inv_n - mu * mu;
  const float rstd = rsqrtf(var + 1e-5f);

  float* op = out + (size_t)p * nout;
  e = 0;
  for (int i = tid; i < nout; i += 256, ++e) {
    const int co = i / (H * W);
    const float v = (vals[e] - mu) * rstd * gw[co] + gb[co];
    op[i] = v > 0.f ? v : 0.f;
  }
}

// ---------------------------------------------------------------------------
// Orchestration
// ---------------------------------------------------------------------------
extern "C" void kernel_launch(void* const* d_in, const int* in_sizes, int n_in,
                              void* d_out, int out_size, void* d_ws, size_t ws_size,
                              hipStream_t stream) {
  (void)in_sizes; (void)n_in; (void)out_size; (void)ws_size;

  const float* query1 = (const float*)d_in[1];   // [4,3,128,32,32]
  const float* query2 = (const float*)d_in[2];   // [4,3,320,32,32]
  const float* query3 = (const float*)d_in[3];   // [4,3,512,32,32]
  const float* supp1  = (const float*)d_in[5];   // [4,1,128,64,64]
  const float* supp2  = (const float*)d_in[6];   // [4,1,320,32,32]
  const float* supp3  = (const float*)d_in[7];   // [4,1,512,16,16]
  const float* q1w = (const float*)d_in[8],  *q1b = (const float*)d_in[9];
  const float* q2w = (const float*)d_in[10], *q2b = (const float*)d_in[11];
  const float* q3w = (const float*)d_in[12], *q3b = (const float*)d_in[13];
  const float* k1w = (const float*)d_in[14], *k1b = (const float*)d_in[15];
  const float* k2w = (const float*)d_in[16], *k2b = (const float*)d_in[17];
  const float* k3w = (const float*)d_in[18], *k3b = (const float*)d_in[19];
  // conv/gn params: e4@20, e3@28, e2@36, e43@44, e32@52 (c_w,c_b,g_w,g_b ×2)

  const int NQ = 3, G = 12, SS = 512, P = 6144;  // P = B*nq*SS

  float* ws = (float*)d_ws;
  size_t off = 0;
  auto A = [&](size_t n) { float* r = ws + off; off += n; return r; };

  float* m0   = A((size_t)P * 256);           // [6144,1,16,16]
  float* m1   = A((size_t)P * 1024);          // [6144,1,32,32]
  float* m2   = A((size_t)P * 4096);          // [6144,1,64,64]
  int*   idxb = (int*)A((size_t)G * SS);
  const size_t mark = off;

  float* qk0 = A((size_t)G * 1024 * 512);
  float* qk1 = A((size_t)G * 1024 * 320);
  float* qk2 = A((size_t)G * 1024 * 128);
  float* sq0 = A((size_t)4 * 256 * 512);
  float* sq1 = A((size_t)4 * 1024 * 320);
  float* sq2 = A((size_t)4 * 4096 * 128);
  float* att = A((size_t)G * 256 * 1024);
  float* sco = A((size_t)G * 1024);

  // ---- projections (x: [G,C,T] -> y: [G,T,O]); 16x64 strip per wave
  proj_gemm_wmma<<<dim3(1024/16, 512/64, G), 32, 0, stream>>>(query3, k3w, k3b, qk0, 512, 1024, 512);
  proj_gemm_wmma<<<dim3(1024/16, 320/64, G), 32, 0, stream>>>(query2, k2w, k2b, qk1, 320, 1024, 320);
  proj_gemm_wmma<<<dim3(1024/16, 128/64, G), 32, 0, stream>>>(query1, k1w, k1b, qk2, 128, 1024, 128);
  proj_gemm_wmma<<<dim3( 256/16, 512/64, 4), 32, 0, stream>>>(supp3,  q3w, q3b, sq0, 512,  256, 512);
  proj_gemm_wmma<<<dim3(1024/16, 320/64, 4), 32, 0, stream>>>(supp2,  q2w, q2b, sq1, 320, 1024, 320);
  proj_gemm_wmma<<<dim3(4096/16, 128/64, 4), 32, 0, stream>>>(supp1,  q1w, q1b, sq2, 128, 4096, 128);

  // ---- full correlation, top-5 pooling, sorted top-512 index selection
  corr_full_wmma<<<dim3(1024/64, 256/16, G), 32, 0, stream>>>(sq0, qk0, att, 256, 1024, 512, NQ);
  topk5_sum_kernel<<<dim3(1024/256, G), 256, 0, stream>>>(att, sco, 256, 1024);
  select_topk_sorted_kernel<<<G, 1024, 0, stream>>>(sco, idxb, 1024, SS);

  // ---- gathered correlations, written directly in maps() layout
  corr_gather_wmma<<<dim3(SS/64,  256/16, G), 32, 0, stream>>>(sq0, qk0, idxb, m0,  256, SS, 512, 1024, NQ);
  corr_gather_wmma<<<dim3(SS/64, 1024/16, G), 32, 0, stream>>>(sq1, qk1, idxb, m1, 1024, SS, 320, 1024, NQ);
  corr_gather_wmma<<<dim3(SS/64, 4096/16, G), 32, 0, stream>>>(sq2, qk2, idxb, m2, 4096, SS, 128, 1024, NQ);

  // ---- conv temporaries reuse the (now dead) projection region
  off = mark;
  float* t4a  = A((size_t)P * 256);           // e4 conv0 out  (1,16,16)
  float* sqz4 = A((size_t)P * 2 * 256);       // e4 out        (2,16,16)
  float* t3a  = A((size_t)P * 1024);          // e3 conv0 out  (1,32,32)
  float* sqz3 = A((size_t)P * 2 * 1024);      // e3 out        (2,32,32)
  float* sqz2 = A((size_t)P * 2 * 4096);      // e2 out        (2,64,64)
  float* m43a = A((size_t)P * 2 * 1024);      // e43 conv0 out (2,32,32)
  float* mx43 = A((size_t)P * 2 * 1024);      // e43 out       (2,32,32)
  float* e32t = A((size_t)P * 2 * 4096);      // e32 conv0 out (2,64,64)
  float* t2a  = (float*)d_out;                // e2 conv0 out borrows d_out

  auto conv = [&](const float* in, const float* in2, int ih2, int iw2,
                  int pi, float* out, int CI, int CO, int H, int W,
                  int KS, int pad) {
    conv_gn_relu_kernel<<<dim3(P), 256, 0, stream>>>(
        in, in2, ih2, iw2,
        (const float*)d_in[pi],     (const float*)d_in[pi + 1],
        (const float*)d_in[pi + 2], (const float*)d_in[pi + 3],
        out, CI, CO, H, W, KS, pad);
  };

  // HPNLearner ('b1')
  conv(m0,   nullptr, 0,  0,  20, t4a,  1, 1, 16, 16, 3, 1);  // e4 c0
  conv(t4a,  nullptr, 0,  0,  24, sqz4, 1, 2, 16, 16, 3, 1);  // e4 c1
  conv(m1,   nullptr, 0,  0,  28, t3a,  1, 1, 32, 32, 3, 1);  // e3 c0
  conv(t3a,  nullptr, 0,  0,  32, sqz3, 1, 2, 32, 32, 3, 1);  // e3 c1
  conv(m2,   nullptr, 0,  0,  36, t2a,  1, 1, 64, 64, 5, 2);  // e2 c0
  conv(t2a,  nullptr, 0,  0,  40, sqz2, 1, 2, 64, 64, 3, 1);  // e2 c1
  // mix43 = e43( resize(sqz4, 32) + sqz3 )   (resize fused into staging)
  conv(sqz3, sqz4,   16, 16,  44, m43a, 2, 2, 32, 32, 3, 1);  // e43 c0
  conv(m43a, nullptr, 0,  0,  48, mx43, 2, 2, 32, 32, 3, 1);  // e43 c1
  // out = e32( resize(mix43, 64) + sqz2 )
  conv(sqz2, mx43,   32, 32,  52, e32t, 2, 2, 64, 64, 3, 1);  // e32 c0
  conv(e32t, nullptr, 0,  0,  56, (float*)d_out, 2, 1, 64, 64, 3, 1); // e32 c1
}
